// TopNAttention_74749610819842
// MI455X (gfx1250) — matrix operand order repaired and checked by reading, not compiled, not run on hardware
//
#include <hip/hip_runtime.h>
#include <hip/hip_bf16.h>

typedef __attribute__((ext_vector_type(2)))  float    v2f;
typedef __attribute__((ext_vector_type(8)))  float    v8f;
typedef __attribute__((ext_vector_type(8)))  _Float16 v8h;
typedef __attribute__((ext_vector_type(16))) _Float16 v16h;

constexpr int BB = 2;      // batch
constexpr int NN = 2048;   // q rows
constexpr int MM = 2048;   // k/v rows
constexpr int DD = 128;    // head dim
constexpr int HH = 8;      // heads
constexpr int DH = 1024;   // DD*HH

constexpr size_t QH16_ELEMS = (size_t)BB * HH * NN * DD;  // 4 Mi f16
constexpr size_t VH_ELEMS   = (size_t)BB * HH * MM * DD;  // 4 Mi f32 (output #1)

// ---------------------------------------------------------------------------
// Projection 1: qproj = q @ Wq + bq  -> qh16 (f16, stored in (b,h,n,d) order)
// One wave computes a 16(row) x 64(col) strip with V_WMMA_F32_16X16X4_F32.
// ---------------------------------------------------------------------------
__global__ __launch_bounds__(128) void proj_q_kernel(
    const float* __restrict__ q, const float* __restrict__ Wq,
    const float* __restrict__ bq, _Float16* __restrict__ qh16)
{
  const int wave = blockIdx.x * 4 + (threadIdx.x >> 5);
  const int lane = threadIdx.x & 31;
  const int half = lane >> 4;
  const int l15  = lane & 15;
  const int mt = wave >> 4;          // 0..255  (BB*NN/16 row tiles)
  const int ns = wave & 15;          // 0..15   (64-col strips)
  const int rowBase = mt * 16;
  const int colBase = ns * 64;

  // A-matrix 16x4 f32 layout: lane -> M = lane&15; VGPR j holds K = 2*half + j
  const float* ar = q + (size_t)(rowBase + l15) * DD;

  v8f acc[4] = {};
  #pragma unroll
  for (int kk = 0; kk < DD; kk += 4) {
    v2f a = *(const v2f*)(ar + kk + 2 * half);
    #pragma unroll
    for (int t = 0; t < 4; ++t) {
      const int c = colBase + t * 16 + l15;
      // B-matrix 4x16 f32: VGPR j holds K = 2*half + j, N = lane&15
      v2f b;
      b.x = Wq[(size_t)(kk + 2 * half    ) * DH + c];
      b.y = Wq[(size_t)(kk + 2 * half + 1) * DH + c];
      acc[t] = __builtin_amdgcn_wmma_f32_16x16x4_f32(
          false, a, false, b, (short)0, acc[t], false, false);
    }
  }

  // C/D 16x16: VGPR j <-> row (j + 8*half), col (lane&15)
  #pragma unroll
  for (int t = 0; t < 4; ++t) {
    const int c    = colBase + t * 16 + l15;
    const float bs = bq[c];
    const int cb = c >> 7, di = c & 127;
    #pragma unroll
    for (int j = 0; j < 8; ++j) {
      const int R  = rowBase + j + 8 * half;   // global GEMM row in [0, BB*NN)
      const int bi = R >> 11;
      const int r  = R & 2047;
      const int hi = r >> 8;
      const int ni = (r & 255) * 8 + cb;       // reshape (n,1024)->(h,n,d)
      qh16[((size_t)(bi * HH + hi) * NN + ni) * DD + di] =
          (_Float16)(acc[t][j] + bs);
    }
  }
}

// ---------------------------------------------------------------------------
// Projection 2: kvproj = concat(k,v) @ Wkv + bkv
//   kh half  -> kh16 (f16 workspace)
//   vh half  -> d_out (f32, output #1)
// ---------------------------------------------------------------------------
__global__ __launch_bounds__(128) void proj_kv_kernel(
    const float* __restrict__ k, const float* __restrict__ v,
    const float* __restrict__ Wkv, const float* __restrict__ bkv,
    _Float16* __restrict__ kh16, float* __restrict__ vh_out)
{
  const int wave = blockIdx.x * 4 + (threadIdx.x >> 5);
  const int lane = threadIdx.x & 31;
  const int half = lane >> 4;
  const int l15  = lane & 15;
  const int mt = wave >> 4;          // 0..511  (BB*2*MM/16 row tiles)
  const int ns = wave & 15;
  const int rowBase = mt * 16;
  const int colBase = ns * 64;

  // Per-lane row gather: per batch, rows [0,2048)=k, [2048,4096)=v
  const int Rl  = rowBase + l15;
  const int bil = Rl >> 12;
  const int rl  = Rl & 4095;
  const float* ar = (rl < MM) ? (k + (size_t)(bil * MM + rl) * DD)
                              : (v + (size_t)(bil * MM + (rl - MM)) * DD);

  v8f acc[4] = {};
  #pragma unroll
  for (int kk = 0; kk < DD; kk += 4) {
    v2f a = *(const v2f*)(ar + kk + 2 * half);
    #pragma unroll
    for (int t = 0; t < 4; ++t) {
      const int c = colBase + t * 16 + l15;
      v2f b;
      b.x = Wkv[(size_t)(kk + 2 * half    ) * DH + c];
      b.y = Wkv[(size_t)(kk + 2 * half + 1) * DH + c];
      acc[t] = __builtin_amdgcn_wmma_f32_16x16x4_f32(
          false, a, false, b, (short)0, acc[t], false, false);
    }
  }

  #pragma unroll
  for (int t = 0; t < 4; ++t) {
    const int c    = colBase + t * 16 + l15;
    const float bs = bkv[c];
    const int cb = c >> 7, di = c & 127;
    #pragma unroll
    for (int j = 0; j < 8; ++j) {
      const int R  = rowBase + j + 8 * half;   // global GEMM row in [0, BB*2*MM)
      const int bi = R >> 12;
      const int r  = R & 4095;
      const int hi = r >> 9;
      const int mi = (r & 511) * 8 + cb;       // reshape (2m,1024)->(h,2m,d)
      const float val = acc[t][j] + bs;
      const size_t base = ((size_t)(bi * HH + hi) * MM);
      if (mi < MM) {
        kh16[(base + mi) * DD + di] = (_Float16)val;        // K half
      } else {
        vh_out[(base + (mi - MM)) * DD + di] = val;         // V half -> output
      }
    }
  }
}

// ---------------------------------------------------------------------------
// Attention: attn[b,n,m] = (1/8) * sum_h relu( qh . kh / sqrt(128) )
// Block = 8 waves computes 128(n) x 64(m).  32 stages (8 heads x 4 K-chunks
// of 32).  Per stage: async-copy A slab (128x32 f16) + B slab (64x32 f16)
// into double-buffered LDS (global_load_async_to_lds_b128 / ASYNCcnt),
// overlap with V_WMMA_F32_16X16X32_F16 on the previous stage.
// ---------------------------------------------------------------------------
__global__ __launch_bounds__(256) void attn_kernel(
    const _Float16* __restrict__ qh16, const _Float16* __restrict__ kh16,
    float* __restrict__ attn_out)
{
  // halves: A buf0 [0,4096) A buf1 [4096,8192) B buf0 [8192,10240) B buf1 [10240,12288)
  __shared__ __align__(32) _Float16 smem[12288];   // 24 KB

  const int tid  = threadIdx.x;
  const int wave = tid >> 5;
  const int lane = tid & 31;
  const int half = lane >> 4;
  const int l15  = lane & 15;

  const int bx  = blockIdx.x;
  const int bi  = bx >> 9;            // 0..1
  const int rem = bx & 511;
  const int nBlock = (rem >> 5) * 128;  // 16 n-blocks
  const int mBlock = (rem & 31) * 64;   // 32 m-blocks

  const unsigned sbase = (unsigned)(uintptr_t)&smem[0];  // LDS byte offset
  const int rpart = tid & 3;           // 16B chunk within a 64B row
  const int row4  = tid >> 2;          // 0..63

  // Issue async copies for stage s: h = s>>2, K-chunk = (s&3)*32, buffer s&1.
  auto issue = [&](int s) {
    const int h   = s >> 2;
    const int kk  = (s & 3) * 32;
    const int buf = s & 1;
    const _Float16* qsrc =
        qh16 + ((size_t)(bi * HH + h) * NN + nBlock) * DD + kk;
    const _Float16* ksrc =
        kh16 + ((size_t)(bi * HH + h) * MM + mBlock) * DD + kk;
    // A slab: 128 rows x 64B  (two 16B transfers per thread)
    const _Float16* g0 = qsrc + (size_t)row4 * DD + rpart * 8;
    unsigned l0 = sbase + buf * 8192u + row4 * 64u + rpart * 16u;
    asm volatile("global_load_async_to_lds_b128 %0, %1, off"
                 :: "v"(l0), "v"(g0) : "memory");
    const _Float16* g1 = g0 + (size_t)64 * DD;
    unsigned l1 = l0 + 64u * 64u;
    asm volatile("global_load_async_to_lds_b128 %0, %1, off"
                 :: "v"(l1), "v"(g1) : "memory");
    // B slab: 64 rows x 64B  (one 16B transfer per thread)
    const _Float16* g2 = ksrc + (size_t)row4 * DD + rpart * 8;
    unsigned l2 = sbase + 16384u + buf * 4096u + row4 * 64u + rpart * 16u;
    asm volatile("global_load_async_to_lds_b128 %0, %1, off"
                 :: "v"(l2), "v"(g2) : "memory");
  };

  const float scale = 0.088388347648318447f;  // 1/sqrt(128)
  const int aRow = wave * 16 + l15;           // this lane's A-slab row
  v8f racc[4] = {};
  v8f cacc[4];

  issue(0);
  #pragma unroll 4
  for (int s = 0; s < 32; ++s) {
    asm volatile("s_wait_asynccnt 0x0" ::: "memory");  // own copies done
    __syncthreads();                                   // => slab[s&1] ready
    if (s + 1 < 32) issue(s + 1);                      // prefetch next stage

    const int aBase = (s & 1) * 4096;          // in halves
    const int bBase = 8192 + (s & 1) * 2048;

    // A frag (16-bit A 16x32): lane half h -> K = h*16+[0,16)
    v16h a = *(const v16h*)&smem[aBase + aRow * 32 + half * 16];

    if ((s & 3) == 0) {
      #pragma unroll
      for (int t = 0; t < 4; ++t) cacc[t] = (v8f)(0.0f);
    }
    #pragma unroll
    for (int t = 0; t < 4; ++t) {
      // B frag (32x16): lane holds kh[mBlock + t*16 + (lane&15)][half*16 + 0..15]
      v16h b = *(const v16h*)&smem[bBase + (t * 16 + l15) * 32 + half * 16];
      cacc[t] = __builtin_amdgcn_wmma_f32_16x16x32_f16(
          false, a, false, b, (short)0, cacc[t], false, false);
    }
    if ((s & 3) == 3) {   // head finished -> relu, accumulate mean
      #pragma unroll
      for (int t = 0; t < 4; ++t)
        #pragma unroll
        for (int j = 0; j < 8; ++j) {
          const float sv = cacc[t][j] * scale;
          racc[t][j] += (sv > 0.0f) ? sv : 0.0f;
        }
    }
  }

  float* outb = attn_out + (size_t)bi * NN * MM;
  #pragma unroll
  for (int t = 0; t < 4; ++t) {
    const int m = mBlock + t * 16 + l15;
    #pragma unroll
    for (int j = 0; j < 8; ++j) {
      const int n = nBlock + wave * 16 + j + 8 * half;
      outb[(size_t)n * MM + m] = racc[t][j] * 0.125f;
    }
  }
}

// ---------------------------------------------------------------------------
extern "C" void kernel_launch(void* const* d_in, const int* in_sizes, int n_in,
                              void* d_out, int out_size, void* d_ws, size_t ws_size,
                              hipStream_t stream) {
  (void)in_sizes; (void)n_in; (void)out_size; (void)ws_size;
  const float* q   = (const float*)d_in[0];
  const float* k   = (const float*)d_in[1];
  const float* v   = (const float*)d_in[2];
  const float* Wq  = (const float*)d_in[3];
  const float* bq  = (const float*)d_in[4];
  const float* Wkv = (const float*)d_in[5];
  const float* bkv = (const float*)d_in[6];

  float* out       = (float*)d_out;
  float* vh_out    = out;              // first BB*HH*MM*DD floats
  float* attn_out  = out + VH_ELEMS;   // then BB*NN*MM floats

  _Float16* qh16 = (_Float16*)d_ws;            // 8 MB
  _Float16* kh16 = qh16 + QH16_ELEMS;          // 8 MB

  // 4096 waves / 4 per block
  proj_q_kernel<<<1024, 128, 0, stream>>>(q, Wq, bq, qh16);
  // 8192 waves
  proj_kv_kernel<<<2048, 128, 0, stream>>>(k, v, Wkv, bkv, kh16, vh_out);
  // 1024 blocks x 8 waves (depends on both, same stream => ordered)
  attn_kernel<<<1024, 256, 0, stream>>>(qh16, kh16, attn_out);
}
